// WaveNet_88416196755968
// MI455X (gfx1250) — matrix operand order repaired and checked by reading, not compiled, not run on hardware
//
#include <hip/hip_runtime.h>
#include <hip/hip_bf16.h>

typedef __attribute__((ext_vector_type(16))) __bf16 v16bf;
typedef __attribute__((ext_vector_type(8)))  float  v8f;

union V16 { v16bf v; __bf16 b[16]; unsigned short s[16]; };

#define TT   8192
#define CC   24
#define BB   32
#define SS   32
#define NBV  256
#define TSV  6146
#define TRIMV 2046
#define LL   20

// manual RNE pack (pack kernel only; hot kernels use hardware __bf16 casts)
__device__ __forceinline__ unsigned short f2bf(float f) {
  unsigned int u = __float_as_uint(f);
  unsigned int r = u + 0x7FFFu + ((u >> 16) & 1u);
  return (unsigned short)(r >> 16);
}
// fast transcendentals on the v_exp_f32 / v_rcp_f32 pipe (co-executes with XDL WMMA)
__device__ __forceinline__ float fsigmoid(float x) {
  return __builtin_amdgcn_rcpf(1.0f + __expf(-x));
}
__device__ __forceinline__ float ftanh(float x) {
  float e = __expf(-2.0f * x);
  return (1.0f - e) * __builtin_amdgcn_rcpf(1.0f + e);
}
__device__ __forceinline__ float eluf(float x) {
  return x > 0.0f ? x : (__expf(x) - 1.0f);
}

// ---------------------------------------------------------------------------
// Pack weights into WMMA A-fragment layout (wave32, 16-bit A 16x32, this code's own
// contiguous k order, not the ISA table): element e of lane l -> row = 16*mtile + (l&15),
// K = 32*ktile + e + ((l>>4)&1)*16
// ---------------------------------------------------------------------------
__global__ void wn_pack(const float* __restrict__ Wsig, const float* __restrict__ Wtanh,
                        const float* __restrict__ Wskip, const float* __restrict__ Wres,
                        const float* __restrict__ bsig, const float* __restrict__ btanh,
                        const float* __restrict__ bskip, const float* __restrict__ bres,
                        const float* __restrict__ W1, const float* __restrict__ W2,
                        unsigned short* __restrict__ A1, unsigned short* __restrict__ A2,
                        unsigned short* __restrict__ W1p, unsigned short* __restrict__ W2p,
                        float* __restrict__ b1p, float* __restrict__ b2p) {
  int gid = blockIdx.x * blockDim.x + threadIdx.x;
  const int R1 = LL * 8 * 32 * 16;   // gate GEMM frags: 4 row-tiles x 2 K-tiles
  const int R2 = LL * 4 * 32 * 16;   // post GEMM frags: 4 row-tiles x 1 K-tile
  const int R3 = 2 * 32 * 16;        // head W1: 2 row-tiles
  const int R4 = 16 * 32 * 16;       // head W2: 16 row-tiles
  const int R5 = LL * 64, R6 = LL * 64;
  if (gid < R1) {
    int e = gid & 15, l = (gid >> 4) & 31, f = (gid >> 9) & 7, i = gid >> 12;
    int mt = f >> 1, kt = f & 1;
    int row = mt * 16 + (l & 15);
    int v = e >> 1, h = e & 1;
    int K = kt * 32 + e + ((l >> 4) & 1) * 16;
    int kr = K & 31, tap = K >> 5;   // K-tile0: x(t-d) tap0, K-tile1: x(t) tap1
    float w = 0.0f;
    if (kr < CC) {
      if (row < CC)                      w = Wsig [((i * CC + row)        * CC + kr) * 2 + tap];
      else if (row >= 32 && row < 32+CC) w = Wtanh[((i * CC + (row - 32)) * CC + kr) * 2 + tap];
    }
    A1[gid] = f2bf(w);
    return;
  }
  gid -= R1;
  if (gid < R2) {
    int e = gid & 15, l = (gid >> 4) & 31, f = (gid >> 9) & 3, i = gid >> 11;
    int row = f * 16 + (l & 15);
    int v = e >> 1, h = e & 1;
    int K = e + ((l >> 4) & 1) * 16;
    float w = 0.0f;
    if (K < CC) {
      if (row < SS)           w = Wskip[(i * SS + row) * CC + K];
      else if (row < 32 + CC) w = Wres [(i * CC + (row - 32)) * CC + K];
    }
    A2[gid] = f2bf(w);
    return;
  }
  gid -= R2;
  if (gid < R3) {
    int e = gid & 15, l = (gid >> 4) & 31, f = gid >> 9;
    int row = f * 16 + (l & 15);
    int v = e >> 1, h = e & 1;
    int K = e + ((l >> 4) & 1) * 16;
    W1p[gid] = f2bf(W1[row * SS + K]);
    return;
  }
  gid -= R3;
  if (gid < R4) {
    int e = gid & 15, l = (gid >> 4) & 31, f = gid >> 9;
    int row = f * 16 + (l & 15);
    int v = e >> 1, h = e & 1;
    int K = e + ((l >> 4) & 1) * 16;
    W2p[gid] = f2bf(W2[row * SS + K]);
    return;
  }
  gid -= R4;
  if (gid < R5) {
    int j = gid & 63, i = gid >> 6;
    float w = 0.0f;
    if (j < CC) w = bsig[i * CC + j];
    else if (j >= 32 && j < 32 + CC) w = btanh[i * CC + (j - 32)];
    b1p[gid] = w;
    return;
  }
  gid -= R5;
  if (gid < R6) {
    int j = gid & 63, i = gid >> 6;
    float w = 0.0f;
    if (j < SS) w = bskip[i * SS + j];
    else if (j < 32 + CC) w = bres[i * CC + (j - 32)];
    b2p[gid] = w;
    return;
  }
}

// ---------------------------------------------------------------------------
// One WaveNet layer: fused gate GEMM (64x64 bf16, K-split over [x(t-d);x(t)]),
// gated activation, fused skip+res GEMM (64x32), skip accumulation, residual.
// One wave per (batch, 16-timestep tile). 8 waves / block, independent.
// ---------------------------------------------------------------------------
__global__ __launch_bounds__(256) void wn_layer(
    const float* __restrict__ xin, float* __restrict__ xout,
    float* __restrict__ skip,
    const unsigned short* __restrict__ A1, const unsigned short* __restrict__ A2,
    const float* __restrict__ bias1, const float* __restrict__ bias2,
    int d, int first, int last) {
  __shared__ __bf16 ush[8][32][17];
  int wave = threadIdx.x >> 5;
  int lane = threadIdx.x & 31;
  int tile = blockIdx.x * 8 + wave;
  int b = blockIdx.y;
  int n = lane & 15, hi = lane >> 4;
  int tc = tile * 16 + n;
  const float* xb = xin + (size_t)b * CC * TT;

  // B fragments (32xK=16 columns): K-tile0 = x(t-d), K-tile1 = x(t)
  V16 B0, B1;
#pragma unroll
  for (int e = 0; e < 16; ++e) {
    int kk = e + 16 * hi;
    float v0 = 0.0f, v1 = 0.0f;
    if (kk < CC) {
      int tm = tc - d; if (tm < 0) tm = 0;
      v0 = xb[kk * TT + tm];
      v1 = xb[kk * TT + tc];
    }
    B0.b[e] = (__bf16)v0;   // hardware f32->bf16 convert
    B1.b[e] = (__bf16)v1;
  }

  const v16bf* a1p = (const v16bf*)A1;
  v8f acc[4];
#pragma unroll
  for (int m = 0; m < 4; ++m) {
    v8f c;
#pragma unroll
    for (int r = 0; r < 8; ++r) c[r] = bias1[16 * m + 8 * hi + r];
    v16bf a0 = a1p[(m * 2 + 0) * 32 + lane];
    v16bf a1 = a1p[(m * 2 + 1) * 32 + lane];
    c = __builtin_amdgcn_wmma_f32_16x16x32_bf16(false, a0, false, B0.v, (short)0, c, false, false);
    c = __builtin_amdgcn_wmma_f32_16x16x32_bf16(false, a1, false, B1.v, (short)0, c, false, false);
    acc[m] = c;
  }

  // u = sigmoid(gs) * tanh(gt); transpose through per-wave LDS to B layout
#pragma unroll
  for (int m = 0; m < 2; ++m)
#pragma unroll
    for (int r = 0; r < 8; ++r) {
      float u = fsigmoid(acc[m][r]) * ftanh(acc[m + 2][r]);
      ush[wave][m * 16 + 8 * hi + r][n] = (__bf16)u;
    }
  V16 Bu;
#pragma unroll
  for (int e = 0; e < 16; ++e) Bu.b[e] = ush[wave][e + 16 * hi][n];

  const v16bf* a2p = (const v16bf*)A2;
  v8f out[4];
#pragma unroll
  for (int m = 0; m < 4; ++m) {
    v8f c;
#pragma unroll
    for (int r = 0; r < 8; ++r) c[r] = bias2[16 * m + 8 * hi + r];
    v16bf a = a2p[m * 32 + lane];
    c = __builtin_amdgcn_wmma_f32_16x16x32_bf16(false, a, false, Bu.v, (short)0, c, false, false);
    out[m] = c;
  }

  // skip accumulation: reference keeps the LAST 6146 samples of every layer
  if (tc >= TRIMV) {
    float* sp = skip + (size_t)b * SS * TSV + (tc - TRIMV);
#pragma unroll
    for (int m = 0; m < 2; ++m)
#pragma unroll
      for (int r = 0; r < 8; ++r) {
        int row = m * 16 + 8 * hi + r;
        if (first) sp[(size_t)row * TSV] = out[m][r];
        else       sp[(size_t)row * TSV] += out[m][r];
      }
  }

  // residual: new_out = W_res@u + x(t)
  if (!last) {
    float* xo = xout + (size_t)b * CC * TT;
#pragma unroll
    for (int m = 2; m < 4; ++m)
#pragma unroll
      for (int r = 0; r < 8; ++r) {
        int ch = m * 16 + 8 * hi + r - 32;
        if (ch < CC) xo[ch * TT + tc] = out[m][r] + xb[ch * TT + tc];
      }
  }
}

// ---------------------------------------------------------------------------
// Head: y = W2 @ elu(W1 @ elu(skip_sum) + b1) + b2, output (B, T', NB)
// Output is write-once (201 MB > 192 MB L2): stream with non-temporal stores
// so the re-read skip/activation buffers stay L2-resident.
// ---------------------------------------------------------------------------
__global__ __launch_bounds__(256) void wn_head(
    const float* __restrict__ skip,
    const unsigned short* __restrict__ W1p, const unsigned short* __restrict__ W2p,
    const float* __restrict__ b1, const float* __restrict__ b2,
    float* __restrict__ out) {
  __shared__ __bf16 zsh[8][32][17];
  int wave = threadIdx.x >> 5;
  int lane = threadIdx.x & 31;
  int tile = blockIdx.x * 8 + wave;
  if (tile * 16 >= TSV) return;  // wave-uniform exit
  int b = blockIdx.y;
  int n = lane & 15, hi = lane >> 4;
  int tc = tile * 16 + n;
  bool valid = tc < TSV;

  V16 Bh;
#pragma unroll
  for (int e = 0; e < 16; ++e) {
    int k = e + 16 * hi;
    float v = 0.0f;
    if (valid) v = eluf(skip[((size_t)b * SS + k) * TSV + tc]);
    Bh.b[e] = (__bf16)v;
  }

  const v16bf* w1 = (const v16bf*)W1p;
#pragma unroll
  for (int m = 0; m < 2; ++m) {
    v8f c;
#pragma unroll
    for (int r = 0; r < 8; ++r) c[r] = b1[16 * m + 8 * hi + r];
    c = __builtin_amdgcn_wmma_f32_16x16x32_bf16(false, w1[m * 32 + lane], false, Bh.v, (short)0, c, false, false);
#pragma unroll
    for (int r = 0; r < 8; ++r)
      zsh[wave][m * 16 + 8 * hi + r][n] = (__bf16)eluf(c[r]);
  }
  V16 Bz;
#pragma unroll
  for (int e = 0; e < 16; ++e) Bz.b[e] = zsh[wave][e + 16 * hi][n];

  const v16bf* w2 = (const v16bf*)W2p;
  float* ob = out + ((size_t)b * TSV + (valid ? tc : 0)) * NBV;
#pragma unroll
  for (int m = 0; m < 16; ++m) {
    v8f c;
#pragma unroll
    for (int r = 0; r < 8; ++r) c[r] = b2[16 * m + 8 * hi + r];
    c = __builtin_amdgcn_wmma_f32_16x16x32_bf16(false, w2[m * 32 + lane], false, Bz.v, (short)0, c, false, false);
    if (valid) {
#pragma unroll
      for (int r = 0; r < 8; ++r)
        __builtin_nontemporal_store(c[r], &ob[16 * m + 8 * hi + r]);
    }
  }
}

extern "C" void kernel_launch(void* const* d_in, const int* in_sizes, int n_in,
                              void* d_out, int out_size, void* d_ws, size_t ws_size,
                              hipStream_t stream) {
  (void)in_sizes; (void)n_in; (void)out_size; (void)ws_size;
  const float* x     = (const float*)d_in[0];
  const float* Wsig  = (const float*)d_in[1];
  const float* bsig  = (const float*)d_in[2];
  const float* Wtanh = (const float*)d_in[3];
  const float* btanh = (const float*)d_in[4];
  const float* Wskip = (const float*)d_in[5];
  const float* bskip = (const float*)d_in[6];
  const float* Wres  = (const float*)d_in[7];
  const float* bres  = (const float*)d_in[8];
  const float* W1    = (const float*)d_in[9];
  const float* b1    = (const float*)d_in[10];
  const float* W2    = (const float*)d_in[11];
  const float* b2    = (const float*)d_in[12];
  float* out = (float*)d_out;

  // workspace carving
  const size_t XB = (size_t)BB * CC * TT * sizeof(float);   // 25.2 MB
  const size_t SK = (size_t)BB * SS * TSV * sizeof(float);  // 25.2 MB
  char* ws = (char*)d_ws;
  float* bufA = (float*)(ws);
  float* bufB = (float*)(ws + XB);
  float* skip = (float*)(ws + 2 * XB);
  unsigned short* A1  = (unsigned short*)(ws + 2 * XB + SK);
  unsigned short* A2  = A1 + LL * 8 * 32 * 16;
  unsigned short* W1p = A2 + LL * 4 * 32 * 16;
  unsigned short* W2p = W1p + 2 * 32 * 16;
  float* b1p = (float*)(W2p + 16 * 32 * 16);
  float* b2p = b1p + LL * 64;

  // pack weights into WMMA fragment layout (cheap, once per launch)
  {
    int total = LL * 8 * 32 * 16 + LL * 4 * 32 * 16 + 2 * 32 * 16 + 16 * 32 * 16 + LL * 64 + LL * 64;
    int blocks = (total + 255) / 256;
    wn_pack<<<blocks, 256, 0, stream>>>(Wsig, Wtanh, Wskip, Wres, bsig, btanh, bskip, bres,
                                        W1, W2, A1, A2, W1p, W2p, b1p, b2p);
  }

  static const int DIL[LL] = {1, 2, 4, 8, 16, 32, 64, 128, 256, 512,
                              1, 2, 4, 8, 16, 32, 64, 128, 256, 512};
  const float* cur = x;
  for (int i = 0; i < LL; ++i) {
    float* nxt = (i & 1) ? bufB : bufA;
    wn_layer<<<dim3(TT / 16 / 8, BB), 256, 0, stream>>>(
        cur, nxt, skip,
        A1 + (size_t)i * 8 * 32 * 16, A2 + (size_t)i * 4 * 32 * 16,
        b1p + i * 64, b2p + i * 64,
        DIL[i], i == 0 ? 1 : 0, i == LL - 1 ? 1 : 0);
    cur = nxt;
  }

  int tilesH = (TSV + 15) / 16;                 // 385
  wn_head<<<dim3((tilesH + 7) / 8, BB), 256, 0, stream>>>(skip, W1p, W2p, b1, b2, out);
}
